// MixFocalTransformerBlock_72971494359723
// MI455X (gfx1250) — matrix operand order repaired and checked by reading, not compiled, not run on hardware
//
#include <hip/hip_runtime.h>

// ---------------------------------------------------------------------------
// MixFocalTransformerBlock for MI455X (gfx1250, wave32, WMMA bf16,
// async global->LDS staging for all GEMMs)
// B=4 T=8 GH=20 GW=36 C=512 HEADS=4 DH=128 WS=(5,9) EXP=(2,4) HD=1960
// ---------------------------------------------------------------------------

typedef __bf16 bf16_t;
typedef __attribute__((ext_vector_type(16))) __bf16 v16bf;
typedef __attribute__((ext_vector_type(8)))  float  v8f;

union Frag16 { v16bf v; unsigned int u[8]; };

#define NTOK   23040          // B*T*GH*GW
#define CCH    512
#define C3     1536
#define NW     64             // B * (GH/5) * (GW/9)
#define NH     4
#define QL     360            // T * 45
#define QLP    368            // padded to 16
#define KL     1320           // 360 + T*120 rolled
#define KLP    1344           // padded to 32
#define DH     128
#define HDIM   1960
#define HDIMP  1984
#define NFRM   32             // B*T
#define FH     60
#define FW     108
#define SCALE_Q 0.08838834764831845f   // 128^-0.5

// -------------------------------- valid-index table ------------------------
__global__ void k_init_valid(int* __restrict__ vg) {
    if (threadIdx.x != 0 || blockIdx.x != 0) return;
    int cnt = 0;
    for (int s = 0; s < 4; ++s) {
        int rlo = (s < 2) ? 0 : 2, rhi = (s < 2) ? 3 : 5;
        int clo = (s & 1) ? 4 : 0, chi = (s & 1) ? 9 : 5;
        for (int i = 0; i < 5; ++i)
            for (int j = 0; j < 9; ++j) {
                bool zeroed = (i >= rlo && i < rhi && j >= clo && j < chi);
                if (!zeroed) vg[cnt++] = s * 45 + i * 9 + j;
            }
    }
}

// -------------------- weight convert: f32 [K][N] -> bf16 [N][Kp] -----------
__global__ void k_wt(const float* __restrict__ src, bf16_t* __restrict__ dst,
                     int K, int N, int Kp) {
    long idx = (long)blockIdx.x * blockDim.x + threadIdx.x;
    long total = (long)N * Kp;
    if (idx >= total) return;
    int kp = (int)(idx % Kp);
    int n  = (int)(idx / Kp);
    float v = (kp < K) ? src[(size_t)kp * N + n] : 0.f;
    dst[(size_t)n * Kp + kp] = (bf16_t)v;
}

// ----------------------------- layernorm -----------------------------------
__global__ void k_layernorm(const float* __restrict__ x,
                            const float* __restrict__ g,
                            const float* __restrict__ b,
                            bf16_t* __restrict__ out, int Mtok) {
    int tok = blockIdx.x * 8 + (threadIdx.x >> 5);
    if (tok >= Mtok) return;
    int lane = threadIdx.x & 31;
    const float* xr = x + (size_t)tok * CCH;
    float vals[16];
    float s = 0.f;
#pragma unroll
    for (int i = 0; i < 16; ++i) { vals[i] = xr[lane * 16 + i]; s += vals[i]; }
#pragma unroll
    for (int o = 16; o >= 1; o >>= 1) s += __shfl_xor(s, o, 32);
    float mean = s * (1.f / 512.f);
    float vs = 0.f;
#pragma unroll
    for (int i = 0; i < 16; ++i) { float d = vals[i] - mean; vs += d * d; }
#pragma unroll
    for (int o = 16; o >= 1; o >>= 1) vs += __shfl_xor(vs, o, 32);
    float rs = rsqrtf(vs * (1.f / 512.f) + 1e-5f);
    bf16_t* orow = out + (size_t)tok * CCH;
#pragma unroll
    for (int i = 0; i < 16; ++i) {
        int c = lane * 16 + i;
        orow[c] = (bf16_t)((vals[i] - mean) * rs * g[c] + b[c]);
    }
}

// --------------------- WMMA GEMM with async LDS staging --------------------
// C[M][N] = A[M][K](bf16, ld=lda) * Bt[N][K](bf16, ld=ldb) + bias (+residual)
// 64x64 block tile, 256 threads (8 waves, 4x2), K multiple of 32.
// A/B k-panels staged into LDS via GLOBAL_LOAD_ASYNC_TO_LDS_B128 (ASYNCcnt).
__global__ void k_gemm_bf16(const bf16_t* __restrict__ A, int lda,
                            const bf16_t* __restrict__ Bt, int ldb,
                            const float* __restrict__ bias,
                            const float* __restrict__ residual,
                            float* __restrict__ outF,
                            bf16_t* __restrict__ outH,
                            int M, int N, int K) {
    __shared__ __align__(16) bf16_t Atile[64 * 32];   // [row][k] 4 KB
    __shared__ __align__(16) bf16_t Btile[64 * 32];   // [col][k] 4 KB

    int bn0 = blockIdx.x * 64;
    int bm0 = blockIdx.y * 64;
    int tid = threadIdx.x;
    int wave = tid >> 5, lane = tid & 31;
    int half = lane >> 4, l16 = lane & 15;
    int wr = wave >> 1;           // 0..3 -> M sub-tile
    int wc = wave & 1;            // 0..1 -> N 32-wide slab

    // per-thread async staging coords: 64 rows x (4 x 16B) per tile
    int srow  = tid >> 2;         // 0..63
    int squad = tid & 3;          // 16B chunk within 64B row
    int a_row = bm0 + srow; if (a_row >= M) a_row = M - 1;
    int b_row = bn0 + srow; if (b_row >= N) b_row = N - 1;
    const bf16_t* gA = A  + (size_t)a_row * lda + squad * 8;
    const bf16_t* gB = Bt + (size_t)b_row * ldb + squad * 8;
    unsigned ldsA = (unsigned)(size_t)&Atile[srow * 32 + squad * 8];
    unsigned ldsB = (unsigned)(size_t)&Btile[srow * 32 + squad * 8];

    const unsigned int* Alds  =
        (const unsigned int*)(Atile + (wr * 16 + l16) * 32);
    const unsigned int* B0lds =
        (const unsigned int*)(Btile + (wc * 32 + l16) * 32);
    const unsigned int* B1lds =
        (const unsigned int*)(Btile + (wc * 32 + 16 + l16) * 32);

    v8f acc0 = {0.f, 0.f, 0.f, 0.f, 0.f, 0.f, 0.f, 0.f};
    v8f acc1 = {0.f, 0.f, 0.f, 0.f, 0.f, 0.f, 0.f, 0.f};

    for (int k0 = 0; k0 < K; k0 += 32) {
        __syncthreads();   // previous k-step's LDS reads complete
        asm volatile("global_load_async_to_lds_b128 %0, %1, off"
                     :: "v"(ldsA),
                        "v"((unsigned long long)(gA + k0)) : "memory");
        asm volatile("global_load_async_to_lds_b128 %0, %1, off"
                     :: "v"(ldsB),
                        "v"((unsigned long long)(gB + k0)) : "memory");
        if (k0 + 32 < K) __builtin_prefetch(gA + k0 + 32, 0, 0);
        asm volatile("s_wait_asynccnt 0" ::: "memory");
        __syncthreads();   // whole tile visible to all waves

        Frag16 a, b0, b1;
#pragma unroll
        for (int v = 0; v < 8; ++v) {
            int ka = ((v >= 4) ? 16 : 0) + half * 8 + ((v & 3) << 1);
            a.u[v]  = Alds[ka >> 1];
            int kb = ((v >= 4) ? 16 : 0) + half * 8 + ((v & 3) << 1);
            b0.u[v] = B0lds[kb >> 1];
            b1.u[v] = B1lds[kb >> 1];
        }
        acc0 = __builtin_amdgcn_wmma_f32_16x16x32_bf16(
            false, a.v, false, b0.v, (short)0, acc0, false, false);
        acc1 = __builtin_amdgcn_wmma_f32_16x16x32_bf16(
            false, a.v, false, b1.v, (short)0, acc1, false, false);
    }

#pragma unroll
    for (int nt = 0; nt < 2; ++nt) {
        const v8f& acc = nt ? acc1 : acc0;
        int n_out = bn0 + wc * 32 + nt * 16 + l16;
        if (n_out >= N) continue;
        float bb = bias ? bias[n_out] : 0.f;
#pragma unroll
        for (int v = 0; v < 8; ++v) {
            int m = bm0 + wr * 16 + v + half * 8;
            if (m < M) {
                size_t off = (size_t)m * N + n_out;
                float val = acc[v] + bb;
                if (residual) val += residual[off];
                if (outF) outF[off] = val;
                else      outH[off] = (bf16_t)val;
            }
        }
    }
}

// --------------------- window / rolled-window gather -----------------------
// qkv: bf16 [NTOK][1536]. Builds:
//   qw   [NW*NH][QLP][128]   (scaled, key-major rows padded with 0)
//   kall [NW*NH][KLP][128]   (key-major, d contiguous)
//   vallT[NW*NH][128][KLP]   (d-major, key contiguous)
__global__ void k_gather(const bf16_t* __restrict__ qkv,
                         const int* __restrict__ vg,
                         bf16_t* __restrict__ qw,
                         bf16_t* __restrict__ kall,
                         bf16_t* __restrict__ vallT) {
    int idx = blockIdx.x * blockDim.x + threadIdx.x;
    int total = NW * NH * KLP;
    if (idx >= total) return;
    int n  = idx % KLP;
    int wh = idx / KLP;
    int h  = wh & 3;
    int w  = wh >> 2;
    int b  = w >> 4, wr = w & 15, wi = wr >> 2, wj = wr & 3;

    bool valid = true;
    int tok = 0;
    if (n < QL) {
        int t = n / 45, pos = n % 45, i = pos / 9, j = pos % 9;
        tok = ((b * 8 + t) * 20 + wi * 5 + i) * 36 + (wj * 9 + j);
    } else if (n < KL) {
        int m = n - QL, t = m / 120, f = m % 120;
        int g = vg[f];
        int s = g / 45, r = g % 45, i = r / 9, j = r % 9;
        int s0 = (s < 2) ? -2 : 2;
        int s1 = (s & 1) ? 4 : -4;
        int gh = ((wi * 5 + i - s0) % 20 + 20) % 20;
        int gw = ((wj * 9 + j - s1) % 36 + 36) % 36;
        tok = ((b * 8 + t) * 20 + gh) * 36 + gw;
    } else {
        valid = false;
    }

    const bf16_t* row = qkv + (size_t)tok * C3 + h * DH;
    bf16_t* krow = kall + ((size_t)wh * KLP + n) * DH;
    for (int d = 0; d < DH; ++d) {
        float kv = valid ? (float)row[CCH + d]     : 0.f;
        float vv = valid ? (float)row[2 * CCH + d] : 0.f;
        krow[d] = (bf16_t)kv;
        vallT[((size_t)wh * DH + d) * KLP + n] = (bf16_t)vv;
        if (n < QLP) {
            float qv = (n < QL) ? (float)row[d] * SCALE_Q : 0.f;
            qw[((size_t)wh * QLP + n) * DH + d] = (bf16_t)qv;
        }
    }
}

// ------------------------------- attention ---------------------------------
// block = (qtile, head, window), 128 threads (4 waves).
// LDS: f32 S[16][KLP] + bf16 P[16][KLP] + f32 rowsum[16]  (~129 KB of 320 KB)
__global__ void k_attention(const bf16_t* __restrict__ qw,
                            const bf16_t* __restrict__ kall,
                            const bf16_t* __restrict__ vallT,
                            bf16_t* __restrict__ proj_in) {
    extern __shared__ char smem[];
    float*  S      = (float*)smem;                                  // 16*KLP f32
    bf16_t* P      = (bf16_t*)(smem + 16 * KLP * 4);                // 16*KLP bf16
    float*  rowsum = (float*)(smem + 16 * KLP * 4 + 16 * KLP * 2);  // 16 f32

    int qt = blockIdx.x, h = blockIdx.y, w = blockIdx.z;
    int wave = threadIdx.x >> 5, lane = threadIdx.x & 31;
    int half = lane >> 4, l16 = lane & 15;
    size_t wh = (size_t)(w * NH + h);

    const bf16_t* Qbase = qw    + (wh * QLP + (size_t)qt * 16) * DH;
    const bf16_t* Kbase = kall  + wh * KLP * DH;
    const bf16_t* Vbase = vallT + wh * DH * KLP;

    // ---- phase 1: S = (Q*scale) @ K^T,  16 x KLP, tiles split over waves
    const unsigned int* Arow = (const unsigned int*)(Qbase + (size_t)l16 * DH);
    for (int kt = wave; kt < KLP / 16; kt += 4) {
        const unsigned int* Brow =
            (const unsigned int*)(Kbase + (size_t)(kt * 16 + l16) * DH);
        v8f acc = {0.f, 0.f, 0.f, 0.f, 0.f, 0.f, 0.f, 0.f};
#pragma unroll
        for (int d0 = 0; d0 < DH; d0 += 32) {
            Frag16 a, bB;
#pragma unroll
            for (int v = 0; v < 8; ++v) {
                int ka = d0 + ((v >= 4) ? 16 : 0) + half * 8 + ((v & 3) << 1);
                a.u[v] = Arow[ka >> 1];
                int kb = d0 + ((v >= 4) ? 16 : 0) + half * 8 + ((v & 3) << 1);
                bB.u[v] = Brow[kb >> 1];
            }
            acc = __builtin_amdgcn_wmma_f32_16x16x32_bf16(
                false, a.v, false, bB.v, (short)0, acc, false, false);
        }
        int col = kt * 16 + l16;
        bool pad = (col >= KL);
#pragma unroll
        for (int v = 0; v < 8; ++v)
            S[(v + half * 8) * KLP + col] = pad ? -1e30f : acc[v];
    }
    __syncthreads();

    // ---- phase 2: row softmax (exp stored as bf16; 1/sum applied later)
    if (threadIdx.x < 16) {
        int r = threadIdx.x;
        float mx = -1e30f;
        for (int c = 0; c < KLP; ++c) mx = fmaxf(mx, S[r * KLP + c]);
        float sum = 0.f;
        for (int c = 0; c < KLP; ++c) {
            float e = __expf(S[r * KLP + c] - mx);
            sum += e;
            P[r * KLP + c] = (bf16_t)e;
        }
        rowsum[r] = sum;
    }
    __syncthreads();

    // ---- phase 3: O = P @ V, each wave handles 2 of 8 d-tiles
    const unsigned int* Prow = (const unsigned int*)(P + (size_t)l16 * KLP);
    for (int nt = wave * 2; nt < wave * 2 + 2; ++nt) {
        const unsigned int* Vrow =
            (const unsigned int*)(Vbase + (size_t)(nt * 16 + l16) * KLP);
        v8f acc = {0.f, 0.f, 0.f, 0.f, 0.f, 0.f, 0.f, 0.f};
        for (int k0 = 0; k0 < KLP; k0 += 32) {
            Frag16 a, bB;
#pragma unroll
            for (int v = 0; v < 8; ++v) {
                int ka = k0 + ((v >= 4) ? 16 : 0) + half * 8 + ((v & 3) << 1);
                a.u[v] = Prow[ka >> 1];             // ds_load from LDS
                int kb = k0 + ((v >= 4) ? 16 : 0) + half * 8 + ((v & 3) << 1);
                bB.u[v] = Vrow[kb >> 1];
            }
            acc = __builtin_amdgcn_wmma_f32_16x16x32_bf16(
                false, a.v, false, bB.v, (short)0, acc, false, false);
        }
        int dcol = nt * 16 + l16;
#pragma unroll
        for (int v = 0; v < 8; ++v) {
            int m = v + half * 8;
            int q = qt * 16 + m;
            if (q < QL) {
                float o = acc[v] / rowsum[m];
                proj_in[((size_t)w * QL + q) * CCH + h * DH + dcol] = (bf16_t)o;
            }
        }
    }
}

// -------------------- attention residual + window reverse ------------------
__global__ void k_attn_res(const float* __restrict__ x,
                           const float* __restrict__ proj_out,
                           float* __restrict__ x1) {
    long idx = (long)blockIdx.x * blockDim.x + threadIdx.x;
    long total = (long)NW * QL * CCH;
    if (idx >= total) return;
    int c  = (int)(idx & (CCH - 1));
    long wt = idx >> 9;
    int w = (int)(wt / QL), n = (int)(wt % QL);
    int b = w >> 4, wr = w & 15, wi = wr >> 2, wj = wr & 3;
    int t = n / 45, pos = n % 45, i = pos / 9, j = pos % 9;
    size_t tok = ((size_t)(b * 8 + t) * 20 + wi * 5 + i) * 36 + (wj * 9 + j);
    x1[tok * CCH + c] = x[tok * CCH + c] + proj_out[(size_t)wt * CCH + c];
}

// ------------------------------- fold --------------------------------------
// folded[f][c40][h][w] = (sum over coinciding (kh,oh),(kw,ow)) / overlapCount
__global__ void k_fold(const float* __restrict__ fc1,
                       float* __restrict__ folded) {
    long idx = (long)blockIdx.x * blockDim.x + threadIdx.x;
    long total = (long)NFRM * 40 * FH * FW;
    if (idx >= total) return;
    int wc = (int)(idx % FW); long rest = idx / FW;
    int hr = (int)(rest % FH); rest /= FH;
    int c40 = (int)(rest % 40);
    int f = (int)(rest / 40);
    int hp = hr + 3, wp = wc + 3;
    int cH = 0, cW = 0;
#pragma unroll
    for (int k = 0; k < 7; ++k) {
        int dh = hp - k;
        if (dh >= 0 && dh % 3 == 0 && dh / 3 < 20) ++cH;
        int dw = wp - k;
        if (dw >= 0 && dw % 3 == 0 && dw / 3 < 36) ++cW;
    }
    float acc = 0.f;
    for (int kh = 0; kh < 7; ++kh) {
        int dh = hp - kh;
        if (dh < 0 || dh % 3 != 0) continue;
        int oh = dh / 3; if (oh >= 20) continue;
        for (int kw = 0; kw < 7; ++kw) {
            int dw = wp - kw;
            if (dw < 0 || dw % 3 != 0) continue;
            int ow = dw / 3; if (ow >= 36) continue;
            acc += fc1[((size_t)f * 720 + oh * 36 + ow) * HDIM +
                       c40 * 49 + kh * 7 + kw];
        }
    }
    folded[idx] = acc / (float)(cH * cW);
}

// ----------------- fused unfold + depthwise conv + GELU --------------------
__global__ void k_dwconv_gelu(const float* __restrict__ folded,
                              const float* __restrict__ w3,
                              const float* __restrict__ b3,
                              const float* __restrict__ w5,
                              const float* __restrict__ b5,
                              bf16_t* __restrict__ out) {
    long idx = (long)blockIdx.x * blockDim.x + threadIdx.x;
    long total = (long)NFRM * HDIMP * 720;
    if (idx >= total) return;
    int sp = (int)(idx % 720); long rest = idx / 720;
    int ch = (int)(rest % HDIMP);
    int f  = (int)(rest / HDIMP);
    size_t ooff = ((size_t)f * 720 + sp) * HDIMP + ch;
    if (ch >= HDIM) { out[ooff] = (bf16_t)0.f; return; }
    int gh = sp / 36, gw = sp % 36;
    int c40 = ch / 49, kk = ch % 49, kh = kk / 7, kw = kk % 7;
    const float* fb = folded + ((size_t)f * 40 + c40) * FH * FW;

    float acc;
    if (ch < 980) {
        acc = b3[ch];
        const float* wt = w3 + (size_t)ch * 9;
        for (int dy = 0; dy < 3; ++dy)
            for (int dx = 0; dx < 3; ++dx) {
                int y = gh + dy - 1, x = gw + dx - 1;
                float u = 0.f;
                if (y >= 0 && y < 20 && x >= 0 && x < 36) {
                    int fh = kh + 3 * y - 3, fw = kw + 3 * x - 3;
                    if (fh >= 0 && fh < FH && fw >= 0 && fw < FW)
                        u = fb[fh * FW + fw];
                }
                acc += wt[dy * 3 + dx] * u;
            }
    } else {
        int c5 = ch - 980;
        acc = b5[c5];
        const float* wt = w5 + (size_t)c5 * 25;
        for (int dy = 0; dy < 5; ++dy)
            for (int dx = 0; dx < 5; ++dx) {
                int y = gh + dy - 2, x = gw + dx - 2;
                float u = 0.f;
                if (y >= 0 && y < 20 && x >= 0 && x < 36) {
                    int fh = kh + 3 * y - 3, fw = kw + 3 * x - 3;
                    if (fh >= 0 && fh < FH && fw >= 0 && fw < FW)
                        u = fb[fh * FW + fw];
                }
                acc += wt[dy * 5 + dx] * u;
            }
    }
    float ge = 0.5f * acc * (1.f + erff(acc * 0.70710678118f));
    out[ooff] = (bf16_t)ge;
}

// ---------------------------------------------------------------------------
extern "C" void kernel_launch(void* const* d_in, const int* in_sizes, int n_in,
                              void* d_out, int out_size, void* d_ws, size_t ws_size,
                              hipStream_t stream) {
    const float* x      = (const float*)d_in[0];
    const float* ln1_g  = (const float*)d_in[1];
    const float* ln1_b  = (const float*)d_in[2];
    const float* ln2_g  = (const float*)d_in[3];
    const float* ln2_b  = (const float*)d_in[4];
    const float* w_qkv  = (const float*)d_in[5];
    const float* b_qkv  = (const float*)d_in[6];
    const float* w_proj = (const float*)d_in[7];
    const float* b_proj = (const float*)d_in[8];
    const float* w_fc1  = (const float*)d_in[9];
    const float* b_fc1  = (const float*)d_in[10];
    const float* w_c3   = (const float*)d_in[11];
    const float* b_c3   = (const float*)d_in[12];
    const float* w_c5   = (const float*)d_in[13];
    const float* b_c5   = (const float*)d_in[14];
    const float* w_fc2  = (const float*)d_in[15];
    const float* b_fc2  = (const float*)d_in[16];
    float* out = (float*)d_out;

    // ---- workspace carve-out (256B aligned slabs) ----
    char* base = (char*)d_ws;
    size_t off = 0;
    auto alloc = [&](size_t bytes) -> char* {
        char* p = base + off;
        off = (off + bytes + 255) & ~(size_t)255;
        return p;
    };
    int*    validg   = (int*)alloc(120 * 4);
    bf16_t* wqkv_t   = (bf16_t*)alloc((size_t)C3 * CCH * 2);
    bf16_t* wproj_t  = (bf16_t*)alloc((size_t)CCH * CCH * 2);
    bf16_t* wfc1_t   = (bf16_t*)alloc((size_t)HDIM * CCH * 2);
    bf16_t* wfc2_t   = (bf16_t*)alloc((size_t)CCH * HDIMP * 2);
    bf16_t* ln_buf   = (bf16_t*)alloc((size_t)NTOK * CCH * 2);
    bf16_t* qkv      = (bf16_t*)alloc((size_t)NTOK * C3 * 2);
    bf16_t* qw       = (bf16_t*)alloc((size_t)NW * NH * QLP * DH * 2);
    bf16_t* kall     = (bf16_t*)alloc((size_t)NW * NH * KLP * DH * 2);
    bf16_t* vallT    = (bf16_t*)alloc((size_t)NW * NH * DH * KLP * 2);
    bf16_t* proj_in  = (bf16_t*)alloc((size_t)NTOK * CCH * 2);
    float*  proj_out = (float*)alloc((size_t)NTOK * CCH * 4);
    float*  x1       = (float*)alloc((size_t)NTOK * CCH * 4);
    float*  fc1_out  = (float*)alloc((size_t)NTOK * HDIM * 4);
    float*  folded   = (float*)alloc((size_t)NFRM * 40 * FH * FW * 4);
    bf16_t* ffn_mid  = (bf16_t*)alloc((size_t)NTOK * HDIMP * 2);

    auto cdiv = [](long a, long b) { return (unsigned)((a + b - 1) / b); };
    auto gemm = [&](const bf16_t* A, int lda, const bf16_t* Bt, int ldb,
                    const float* bias, const float* res,
                    float* oF, bf16_t* oH, int M, int N, int K) {
        dim3 grid(cdiv(N, 64), cdiv(M, 64));
        k_gemm_bf16<<<grid, 256, 0, stream>>>(
            A, lda, Bt, ldb, bias, res, oF, oH, M, N, K);
    };

    // 0. valid-roll index table + weight conversion
    k_init_valid<<<1, 1, 0, stream>>>(validg);
    k_wt<<<cdiv((long)C3 * CCH, 256), 256, 0, stream>>>(w_qkv, wqkv_t, CCH, C3, CCH);
    k_wt<<<cdiv((long)CCH * CCH, 256), 256, 0, stream>>>(w_proj, wproj_t, CCH, CCH, CCH);
    k_wt<<<cdiv((long)HDIM * CCH, 256), 256, 0, stream>>>(w_fc1, wfc1_t, CCH, HDIM, CCH);
    k_wt<<<cdiv((long)CCH * HDIMP, 256), 256, 0, stream>>>(w_fc2, wfc2_t, HDIM, CCH, HDIMP);

    // 1. LN1 -> bf16
    k_layernorm<<<cdiv(NTOK, 8), 256, 0, stream>>>(x, ln1_g, ln1_b, ln_buf, NTOK);

    // 2. qkv GEMM (bf16 out)
    gemm(ln_buf, CCH, wqkv_t, CCH, b_qkv, nullptr, nullptr, qkv, NTOK, C3, CCH);

    // 3. window / rolled gather
    k_gather<<<cdiv((long)NW * NH * KLP, 256), 256, 0, stream>>>(
        qkv, validg, qw, kall, vallT);

    // 4. attention (flash-in-LDS, WMMA both matmuls)
    {
        dim3 grid(QLP / 16, NH, NW);
        size_t smem = (size_t)16 * KLP * 4 + (size_t)16 * KLP * 2 + 64;
        k_attention<<<grid, 128, smem, stream>>>(qw, kall, vallT, proj_in);
    }

    // 5. proj GEMM + window-reverse residual
    gemm(proj_in, CCH, wproj_t, CCH, b_proj, nullptr, proj_out, nullptr,
         NTOK, CCH, CCH);
    k_attn_res<<<cdiv((long)NW * QL * CCH, 256), 256, 0, stream>>>(x, proj_out, x1);

    // 6. LN2 -> bf16
    k_layernorm<<<cdiv(NTOK, 8), 256, 0, stream>>>(x1, ln2_g, ln2_b, ln_buf, NTOK);

    // 7. fc1 GEMM (f32 out)
    gemm(ln_buf, CCH, wfc1_t, CCH, b_fc1, nullptr, fc1_out, nullptr,
         NTOK, HDIM, CCH);

    // 8. fold(+norm), fused unfold + dwconv + GELU
    k_fold<<<cdiv((long)NFRM * 40 * FH * FW, 256), 256, 0, stream>>>(fc1_out, folded);
    k_dwconv_gelu<<<cdiv((long)NFRM * HDIMP * 720, 256), 256, 0, stream>>>(
        folded, w_c3, b_c3, w_c5, b_c5, ffn_mid);

    // 9. fc2 GEMM + residual -> d_out
    gemm(ffn_mid, HDIMP, wfc2_t, HDIMP, b_fc2, x1, out, nullptr,
         NTOK, CCH, HDIMP);
}
